// SparseAttention_54503134986459
// MI455X (gfx1250) — matrix operand order repaired and checked by reading, not compiled, not run on hardware
//
#include <hip/hip_runtime.h>
#include <hip/hip_bf16.h>

#define B_    2
#define S_    2048
#define D_    1024
#define H_    16
#define HD_   64
#define KKEEP 204   // int(0.1 * 2048)

typedef __bf16 bf16x16 __attribute__((ext_vector_type(16)));
typedef float  f32x8   __attribute__((ext_vector_type(8)));
typedef float  f32x4   __attribute__((ext_vector_type(4)));
typedef unsigned short u16x8 __attribute__((ext_vector_type(8)));
typedef int    v4i     __attribute__((ext_vector_type(4)));

#if __has_builtin(__builtin_amdgcn_global_load_async_to_lds_b128)
#define HAVE_ASYNC_LDS 1
typedef __attribute__((address_space(1))) v4i gv4i;   // global int4
typedef __attribute__((address_space(3))) v4i lv4i;   // LDS int4
#endif

union Frag16 {
  bf16x16 v;
  u16x8   h[2];
  unsigned short u[16];
};

__device__ __forceinline__ unsigned short f2bf(float f) {
  unsigned int u = __float_as_uint(f);
  u += 0x7FFFu + ((u >> 16) & 1u);      // round-to-nearest-even
  return (unsigned short)(u >> 16);
}

__device__ __forceinline__ f32x8 zero8() {
  f32x8 z;
  #pragma unroll
  for (int i = 0; i < 8; ++i) z[i] = 0.0f;
  return z;
}

__device__ __forceinline__ f32x8 wmma_bf16(const Frag16& a, const Frag16& b, f32x8 c) {
  return __builtin_amdgcn_wmma_f32_16x16x32_bf16(false, a.v, false, b.v,
                                                 (short)0, c, false, false);
}

__device__ __forceinline__ void wait_async_then_barrier() {
#if defined(HAVE_ASYNC_LDS)
#if __has_builtin(__builtin_amdgcn_s_wait_asynccnt)
  __builtin_amdgcn_s_wait_asynccnt(0);
#else
  asm volatile("s_wait_asynccnt 0x0" ::: "memory");
#endif
#endif
  __syncthreads();
}

// ---------------------------------------------------------------------------
// Kernel 1: QKV projections.  C[m,n] = A[m,:] @ W[:,n] + bias[n]
// grid = (N/128, M/128, 3); block = 256 (8 waves). 128x128 tile, K-step 32.
// Wave (wr,wc) with wr=wave>>2, wc=wave&3 owns a 64x32 sub-block (4x2 WMMA).
// mode 0/1 -> bf16 [bh][s][hd]; mode 2 (V) -> transposed [bh][hd][s].
// ---------------------------------------------------------------------------
__global__ __launch_bounds__(256)
void sattn_qkv_proj(const float* __restrict__ q_in, const float* __restrict__ k_in,
                    const float* __restrict__ v_in,
                    const float* __restrict__ Wq, const float* __restrict__ bq,
                    const float* __restrict__ Wk, const float* __restrict__ bk,
                    const float* __restrict__ Wv, const float* __restrict__ bv,
                    unsigned short* __restrict__ Qh, unsigned short* __restrict__ Kh,
                    unsigned short* __restrict__ Vt) {
  const int mode = blockIdx.z;
  const float* A    = (mode == 0) ? q_in : (mode == 1) ? k_in : v_in;
  const float* W    = (mode == 0) ? Wq   : (mode == 1) ? Wk   : Wv;
  const float* bias = (mode == 0) ? bq   : (mode == 1) ? bk   : bv;
  unsigned short* out = (mode == 0) ? Qh : (mode == 1) ? Kh : Vt;

  const int n0   = blockIdx.x * 128;
  const int m0   = blockIdx.y * 128;
  const int tid  = threadIdx.x;
  const int lane = tid & 31;
  const int wave = tid >> 5;
  const int wr   = wave >> 2;     // 0..1  (M direction, 64 rows each)
  const int wc   = wave & 3;      // 0..3  (N direction, 32 cols each)

  __shared__ unsigned short lA[128 * 32];   // [m][k] bf16
  __shared__ unsigned short lB[128 * 32];   // [n][k] bf16 (W transposed)

  f32x8 acc[4][2];
  #pragma unroll
  for (int mi = 0; mi < 4; ++mi)
    #pragma unroll
    for (int nj = 0; nj < 2; ++nj) acc[mi][nj] = zero8();

  const int m_l = lane & 15;
  const int kbA = (lane < 16) ? 0 : 8;
  const int n_l = lane & 15;
  const int kbB = (lane < 16) ? 0 : 8;

  for (int k0 = 0; k0 < D_; k0 += 32) {
    __syncthreads();
    {   // stage A: 128x32 fp32 -> bf16
      int r = tid >> 1, sg = (tid & 1) * 16;
      const float* src = A + (size_t)(m0 + r) * D_ + k0 + sg;
      unsigned short* dst = lA + r * 32 + sg;
      #pragma unroll
      for (int i = 0; i < 16; i += 4) {
        f32x4 v = *(const f32x4*)(src + i);
        dst[i + 0] = f2bf(v[0]); dst[i + 1] = f2bf(v[1]);
        dst[i + 2] = f2bf(v[2]); dst[i + 3] = f2bf(v[3]);
      }
      if (k0 + 32 < D_) __builtin_prefetch(src + 32, 0, 1);
    }
    {   // stage W transposed: 32x128 -> [n][k]
      int kr = tid >> 3, sg = (tid & 7) * 16;
      const float* src = W + (size_t)(k0 + kr) * D_ + n0 + sg;
      #pragma unroll
      for (int i = 0; i < 16; ++i)
        lB[(sg + i) * 32 + kr] = f2bf(src[i]);
      if (k0 + 32 < D_) __builtin_prefetch(W + (size_t)(k0 + 32 + kr) * D_ + n0 + sg, 0, 1);
    }
    __syncthreads();

    Frag16 af[4];
    #pragma unroll
    for (int mi = 0; mi < 4; ++mi) {
      const unsigned short* ap = lA + ((wr * 4 + mi) * 16 + m_l) * 32;
      af[mi].h[0] = *(const u16x8*)(ap + kbA);
      af[mi].h[1] = *(const u16x8*)(ap + kbA + 16);
    }
    Frag16 bf[2];
    #pragma unroll
    for (int nj = 0; nj < 2; ++nj) {
      const unsigned short* bp = lB + ((wc * 2 + nj) * 16 + n_l) * 32 + kbB;
      bf[nj].h[0] = *(const u16x8*)(bp);
      bf[nj].h[1] = *(const u16x8*)(bp + 16);
    }
    #pragma unroll
    for (int mi = 0; mi < 4; ++mi)
      #pragma unroll
      for (int nj = 0; nj < 2; ++nj)
        acc[mi][nj] = wmma_bf16(af[mi], bf[nj], acc[mi][nj]);
  }

  // epilogue: bias + scatter into head-split (or transposed, for V) bf16 layout
  const int moff = (lane < 16) ? 0 : 8;
  #pragma unroll
  for (int nj = 0; nj < 2; ++nj) {
    int n = n0 + (wc * 2 + nj) * 16 + n_l;
    float bv_ = bias[n];
    int h = n >> 6, hd = n & 63;
    #pragma unroll
    for (int mi = 0; mi < 4; ++mi) {
      #pragma unroll
      for (int r = 0; r < 8; ++r) {
        int m = m0 + (wr * 4 + mi) * 16 + moff + r;
        int b = m >> 11, s = m & (S_ - 1);
        float val = acc[mi][nj][r] + bv_;
        size_t idx;
        if (mode == 2) idx = (((size_t)(b * H_ + h)) * HD_ + hd) * S_ + s;   // V^T
        else           idx = (((size_t)(b * H_ + h)) * S_ + s) * HD_ + hd;   // Q,K
        out[idx] = f2bf(val);
      }
    }
  }
}

// ---------------------------------------------------------------------------
// Kernel 2: per (bh, 16-query tile) attention with top-k threshold.
// grid = (S/16, B*H); block = 256 (8 waves); dynamic LDS = 16*2048 f32 scores
// (entire score block resident thanks to CDNA5's 320KB LDS) + reduce buffers.
// ---------------------------------------------------------------------------
__global__ __launch_bounds__(256)
void sattn_attention(const unsigned short* __restrict__ Qh,
                     const unsigned short* __restrict__ Kh,
                     const unsigned short* __restrict__ Vt,
                     unsigned short* __restrict__ attnOut) {
  extern __shared__ float smem[];
  float* sc     = smem;                  // [16][2048] scores -> exp weights
  float* red    = smem + 16 * S_;        // [4][256] cross-wave partials
  float* rowsum = red + 4 * 256;         // [16]

  const int q0   = blockIdx.x * 16;
  const int bh   = blockIdx.y;
  const int tid  = threadIdx.x;
  const int lane = tid & 31;
  const int wave = tid >> 5;

  const int m_l  = lane & 15;
  const int kbA  = (lane < 16) ? 0 : 8;
  const int n_l  = lane & 15;
  const int kbB  = (lane < 16) ? 0 : 8;
  const int moff = (lane < 16) ? 0 : 8;

  // ---- phase 1: scores = (Q K^T) / sqrt(HD), tiled 16x16, K = HD = 64 ----
  Frag16 aq0, aq1;
  {
    const unsigned short* qa = Qh + ((size_t)bh * S_ + q0 + m_l) * HD_;
    aq0.h[0] = *(const u16x8*)(qa + kbA);       aq0.h[1] = *(const u16x8*)(qa + kbA + 16);
    aq1.h[0] = *(const u16x8*)(qa + 32 + kbA);  aq1.h[1] = *(const u16x8*)(qa + 32 + kbA + 16);
  }
  for (int kt = wave; kt < S_ / 16; kt += 8) {
    const unsigned short* kp = Kh + ((size_t)bh * S_ + kt * 16 + n_l) * HD_;
    Frag16 b0, b1;
    b0.h[0] = *(const u16x8*)(kp + kbB);       b0.h[1] = *(const u16x8*)(kp + kbB + 16);
    b1.h[0] = *(const u16x8*)(kp + 32 + kbB);  b1.h[1] = *(const u16x8*)(kp + 32 + kbB + 16);
    f32x8 acc = zero8();
    acc = wmma_bf16(aq0, b0, acc);
    acc = wmma_bf16(aq1, b1, acc);
    #pragma unroll
    for (int r = 0; r < 8; ++r)
      sc[(moff + r) * S_ + kt * 16 + n_l] = acc[r] * 0.125f;   // 1/sqrt(64)
  }
  __syncthreads();

  // ---- phase 2: per-row top-k threshold (bisection on count) + softmax ----
  for (int row = wave * 2; row < wave * 2 + 2; ++row) {
    float* rp = sc + row * S_;
    float mx = -3.4e38f, mn = 3.4e38f;
    for (int i = 0; i < 64; ++i) {
      float v = rp[lane + i * 32];
      mx = fmaxf(mx, v); mn = fminf(mn, v);
    }
    #pragma unroll
    for (int off = 16; off; off >>= 1) {
      mx = fmaxf(mx, __shfl_xor(mx, off, 32));
      mn = fminf(mn, __shfl_xor(mn, off, 32));
    }
    // largest t with count(row >= t) >= KKEEP  (== k-th largest value)
    float lo = mn, hi = mx;
    for (int it = 0; it < 24; ++it) {
      float mid = 0.5f * (lo + hi);
      int c = 0;
      for (int i = 0; i < 64; ++i) c += (rp[lane + i * 32] >= mid) ? 1 : 0;
      #pragma unroll
      for (int off = 16; off; off >>= 1) c += __shfl_xor(c, off, 32);
      if (c >= KKEEP) lo = mid; else hi = mid;
    }
    float th = lo;
    float s = 0.0f;
    for (int i = 0; i < 64; ++i) {
      float v = rp[lane + i * 32];
      float e = (v >= th) ? __expf(v - mx) : 0.0f;
      s += e;
      rp[lane + i * 32] = e;     // unnormalized weight
    }
    #pragma unroll
    for (int off = 16; off; off >>= 1) s += __shfl_xor(s, off, 32);
    if (lane == 0) rowsum[row] = s;
  }
  __syncthreads();

  // ---- phase 3: out = weights @ V  (16 x 64, inner = 2048 keys) ----------
  const int j   = wave & 3;
  const int ks0 = (wave < 4) ? 0 : 32;
  f32x8 acc = zero8();
  const unsigned short* vb = Vt + ((size_t)bh * HD_ + j * 16 + n_l) * S_;
  for (int ks = ks0; ks < ks0 + 32; ++ks) {
    Frag16 a;
    const float* ap = sc + m_l * S_ + ks * 32;
    #pragma unroll
    for (int e = 0; e < 8; ++e) a.u[e]     = f2bf(ap[kbA + e]);
    #pragma unroll
    for (int e = 0; e < 8; ++e) a.u[8 + e] = f2bf(ap[kbA + 16 + e]);
    Frag16 b;
    const unsigned short* bp = vb + ks * 32 + kbB;
    b.h[0] = *(const u16x8*)(bp);
    b.h[1] = *(const u16x8*)(bp + 16);
    acc = wmma_bf16(a, b, acc);
  }
  if (wave >= 4) {
    #pragma unroll
    for (int r = 0; r < 8; ++r) red[j * 256 + r * 32 + lane] = acc[r];
  }
  __syncthreads();
  if (wave < 4) {
    const int b_ = bh >> 4, h = bh & 15;
    #pragma unroll
    for (int r = 0; r < 8; ++r) {
      int mrow = moff + r;
      float val = (acc[r] + red[j * 256 + r * 32 + lane]) / rowsum[mrow];
      size_t idx = ((size_t)(b_ * S_ + q0 + mrow)) * D_ + h * HD_ + j * 16 + n_l;
      attnOut[idx] = f2bf(val);
    }
  }
}

// ---------------------------------------------------------------------------
// Kernel 3: output projection.  out = attnOut(bf16) @ Wo + bo  -> fp32
// grid = (N/128, M/128); block = 256.  A-tile staged with ASYNC global->LDS
// copies (ASYNCcnt / s_wait_asynccnt) when the builtin exists.
// ---------------------------------------------------------------------------
__global__ __launch_bounds__(256)
void sattn_out_proj(const unsigned short* __restrict__ Ain,
                    const float* __restrict__ Wo, const float* __restrict__ bo,
                    float* __restrict__ out) {
  const int n0   = blockIdx.x * 128;
  const int m0   = blockIdx.y * 128;
  const int tid  = threadIdx.x;
  const int lane = tid & 31;
  const int wave = tid >> 5;
  const int wr   = wave >> 2;
  const int wc   = wave & 3;

  __shared__ unsigned short lA[128 * 32];
  __shared__ unsigned short lB[128 * 32];

  f32x8 acc[4][2];
  #pragma unroll
  for (int mi = 0; mi < 4; ++mi)
    #pragma unroll
    for (int nj = 0; nj < 2; ++nj) acc[mi][nj] = zero8();

  const int m_l = lane & 15;
  const int kbA = (lane < 16) ? 0 : 8;
  const int n_l = lane & 15;
  const int kbB = (lane < 16) ? 0 : 8;

  for (int k0 = 0; k0 < D_; k0 += 32) {
    __syncthreads();
    {   // stage A (already bf16): async DMA global -> LDS when available
      int r = tid >> 1, sg = (tid & 1) * 16;
      const unsigned short* src = Ain + (size_t)(m0 + r) * D_ + k0 + sg;
      unsigned short* dst = lA + r * 32 + sg;
#if defined(HAVE_ASYNC_LDS)
      __builtin_amdgcn_global_load_async_to_lds_b128(
          (gv4i*)(src), (lv4i*)(dst), 0, 0);
      __builtin_amdgcn_global_load_async_to_lds_b128(
          (gv4i*)(src + 8), (lv4i*)(dst + 8), 0, 0);
#else
      *(u16x8*)(dst)     = *(const u16x8*)(src);
      *(u16x8*)(dst + 8) = *(const u16x8*)(src + 8);
#endif
      if (k0 + 32 < D_) __builtin_prefetch(src + 32, 0, 1);
    }
    {   // stage Wo transposed fp32 -> bf16 [n][k]
      int kr = tid >> 3, sg = (tid & 7) * 16;
      const float* src = Wo + (size_t)(k0 + kr) * D_ + n0 + sg;
      #pragma unroll
      for (int i = 0; i < 16; ++i)
        lB[(sg + i) * 32 + kr] = f2bf(src[i]);
    }
    wait_async_then_barrier();

    Frag16 af[4];
    #pragma unroll
    for (int mi = 0; mi < 4; ++mi) {
      const unsigned short* ap = lA + ((wr * 4 + mi) * 16 + m_l) * 32;
      af[mi].h[0] = *(const u16x8*)(ap + kbA);
      af[mi].h[1] = *(const u16x8*)(ap + kbA + 16);
    }
    Frag16 bf[2];
    #pragma unroll
    for (int nj = 0; nj < 2; ++nj) {
      const unsigned short* bp = lB + ((wc * 2 + nj) * 16 + n_l) * 32 + kbB;
      bf[nj].h[0] = *(const u16x8*)(bp);
      bf[nj].h[1] = *(const u16x8*)(bp + 16);
    }
    #pragma unroll
    for (int mi = 0; mi < 4; ++mi)
      #pragma unroll
      for (int nj = 0; nj < 2; ++nj)
        acc[mi][nj] = wmma_bf16(af[mi], bf[nj], acc[mi][nj]);
  }

  const int moff = (lane < 16) ? 0 : 8;
  #pragma unroll
  for (int nj = 0; nj < 2; ++nj) {
    int n = n0 + (wc * 2 + nj) * 16 + n_l;
    float bv_ = bo[n];
    #pragma unroll
    for (int mi = 0; mi < 4; ++mi) {
      #pragma unroll
      for (int r = 0; r < 8; ++r) {
        int m = m0 + (wr * 4 + mi) * 16 + moff + r;
        out[(size_t)m * D_ + n] = acc[mi][nj][r] + bv_;
      }
    }
  }
}

// ---------------------------------------------------------------------------
extern "C" void kernel_launch(void* const* d_in, const int* in_sizes, int n_in,
                              void* d_out, int out_size, void* d_ws, size_t ws_size,
                              hipStream_t stream) {
  const float* q_in = (const float*)d_in[0];
  const float* k_in = (const float*)d_in[1];
  const float* v_in = (const float*)d_in[2];
  const float* Wq   = (const float*)d_in[3];
  const float* bq   = (const float*)d_in[4];
  const float* Wk   = (const float*)d_in[5];
  const float* bk   = (const float*)d_in[6];
  const float* Wv   = (const float*)d_in[7];
  const float* bv   = (const float*)d_in[8];
  const float* Wo   = (const float*)d_in[9];
  const float* bo   = (const float*)d_in[10];

  const size_t HEAD_ELEMS = (size_t)B_ * H_ * S_ * HD_;   // 4,194,304
  unsigned short* ws = (unsigned short*)d_ws;
  unsigned short* Qh = ws;
  unsigned short* Kh = ws + HEAD_ELEMS;
  unsigned short* Vt = ws + 2 * HEAD_ELEMS;
  unsigned short* aO = ws + 3 * HEAD_ELEMS;

  // 1) QKV projections (z = 0:Q, 1:K, 2:V-transposed), 128x128 tiles
  sattn_qkv_proj<<<dim3(D_ / 128, (B_ * S_) / 128, 3), dim3(256), 0, stream>>>(
      q_in, k_in, v_in, Wq, bq, Wk, bk, Wv, bv, Qh, Kh, Vt);

  // 2) attention with top-k sparsification (128 q-tiles x 32 bh)
  size_t smem = (size_t)(16 * S_ + 4 * 256 + 16) * sizeof(float);   // 135,232 B
  (void)hipFuncSetAttribute(reinterpret_cast<const void*>(&sattn_attention),
                            hipFuncAttributeMaxDynamicSharedMemorySize, (int)smem);
  sattn_attention<<<dim3(S_ / 16, B_ * H_), dim3(256), smem, stream>>>(Qh, Kh, Vt, aO);

  // 3) output projection -> fp32 d_out
  sattn_out_proj<<<dim3(D_ / 128, (B_ * S_) / 128), dim3(256), 0, stream>>>(
      aO, Wo, bo, (float*)d_out);
}
